// InfoNCE_Loss_69750268887633
// MI455X (gfx1250) — matrix operand order repaired and checked by reading, not compiled, not run on hardware
//
#include <hip/hip_runtime.h>
#include <hip/hip_bf16.h>
#include <math.h>

typedef _Float16 half_t;
typedef __attribute__((ext_vector_type(16))) _Float16 v16h;
typedef __attribute__((ext_vector_type(8)))  _Float16 v8h;
typedef __attribute__((ext_vector_type(4)))  _Float16 v4h;
typedef __attribute__((ext_vector_type(8)))  float    v8f;
typedef __attribute__((ext_vector_type(4)))  float    v4f;

#define N_ROWS 8192
#define KDIM   512
#define INV_T  10.0f   // 1 / TEMPERATURE

#define BM 128   // block tile M  (2 waves * 64)
#define BN 256   // block tile N  (4 waves * 64)
#define WM 64    // per-wave M (4 x 16)
#define WN 64    // per-wave N (4 x 16)

// ---------------------------------------------------------------------------
// Row-wise L2 normalize f32 -> f16.  One wave32 per row, 8 rows per block.
// ---------------------------------------------------------------------------
__global__ void __launch_bounds__(256)
nrm_rows_f16(const float* __restrict__ x, half_t* __restrict__ y) {
    const int wave = threadIdx.x >> 5;
    const int lane = threadIdx.x & 31;
    const int row  = blockIdx.x * 8 + wave;

    const float* xr = x + (size_t)row * KDIM;
    half_t*      yr = y + (size_t)row * KDIM;

    v4f v[4];
#pragma unroll
    for (int i = 0; i < 4; ++i)
        v[i] = *(const v4f*)(xr + (i * 32 + lane) * 4);

    float ss = 0.0f;
#pragma unroll
    for (int i = 0; i < 4; ++i)
#pragma unroll
        for (int j = 0; j < 4; ++j)
            ss += v[i][j] * v[i][j];

#pragma unroll
    for (int off = 16; off > 0; off >>= 1)
        ss += __shfl_xor(ss, off, 32);

    const float inv = 1.0f / fmaxf(sqrtf(ss), 1e-12f);

#pragma unroll
    for (int i = 0; i < 4; ++i) {
        v4h h;
#pragma unroll
        for (int j = 0; j < 4; ++j)
            h[j] = (half_t)(v[i][j] * inv);
        *(v4h*)(yr + (i * 32 + lane) * 4) = h;
    }
}

// ---------------------------------------------------------------------------
// Fused sim = A * B^T (rows of B are columns of sim), exp(sim/T), global sum.
// A, B are row-major f16 [8192][512] unit vectors.  Each wave computes a
// 64x64 output tile as 4x4 v_wmma_f32_16x16x32_f16 fragments, then reduces
// exp() of every element into a single double via one atomic per wave.
// Fragment layouts follow CDNA5 ISA 7.12.2 (wave32).
// ---------------------------------------------------------------------------
__global__ void __launch_bounds__(256)
simexp_gemm(const half_t* __restrict__ A, const half_t* __restrict__ B,
            double* __restrict__ accum) {
    const int tid   = threadIdx.x;
    const int wave  = tid >> 5;
    const int lane  = tid & 31;
    const int waveM = wave >> 2;     // 0..1
    const int waveN = wave & 3;      // 0..3

    const int mBase   = blockIdx.x * BM + waveM * WM;
    const int nBase   = blockIdx.y * BN + waveN * WN;
    const int laneLo  = lane & 15;
    const int halfSel = lane >> 4;   // 0 for lanes 0-15, 1 for lanes 16-31

    const half_t* aRow[4];
    const half_t* bRow[4];
#pragma unroll
    for (int i = 0; i < 4; ++i) {
        aRow[i] = A + (size_t)(mBase + i * 16 + laneLo) * KDIM;
        bRow[i] = B + (size_t)(nBase + i * 16 + laneLo) * KDIM;
    }
    const int aOff = halfSel * 8;    // hi half-wave holds K+8..15 / K+24..31
    const int bOff = halfSel * 8;    // hi half-wave holds K+8..15 / K+24..31

    v8f acc[4][4] = {};

    for (int k = 0; k < KDIM; k += 32) {
        if (k + 32 < KDIM) {         // warm L0 for the next K-slice
#pragma unroll
            for (int i = 0; i < 4; ++i) {
                __builtin_prefetch(aRow[i] + k + 32 + aOff, 0, 3);
                __builtin_prefetch(bRow[i] + k + 32 + bOff, 0, 3);
            }
        }

        v16h afrag[4], bfrag[4];
#pragma unroll
        for (int i = 0; i < 4; ++i) {
            v8h lo = *(const v8h*)(aRow[i] + k + aOff);
            v8h hi = *(const v8h*)(aRow[i] + k + 16 + aOff);
            afrag[i] = __builtin_shufflevector(lo, hi,
                        0, 1, 2, 3, 4, 5, 6, 7, 8, 9, 10, 11, 12, 13, 14, 15);
            v8h blo = *(const v8h*)(bRow[i] + k + bOff);
            v8h bhi = *(const v8h*)(bRow[i] + k + 16 + bOff);
            bfrag[i] = __builtin_shufflevector(blo, bhi,
                        0, 1, 2, 3, 4, 5, 6, 7, 8, 9, 10, 11, 12, 13, 14, 15);
        }

#pragma unroll
        for (int mi = 0; mi < 4; ++mi)
#pragma unroll
            for (int ni = 0; ni < 4; ++ni)
                acc[mi][ni] = __builtin_amdgcn_wmma_f32_16x16x32_f16(
                    false, afrag[mi], false, bfrag[ni],
                    (short)0, acc[mi][ni], false, false);
    }

    // exp(sim / T) and full-tile sum; the total sum is invariant to the C/D
    // VGPR layout, so no deswizzle is needed.
    float s = 0.0f;
#pragma unroll
    for (int mi = 0; mi < 4; ++mi)
#pragma unroll
        for (int ni = 0; ni < 4; ++ni)
#pragma unroll
            for (int e = 0; e < 8; ++e)
                s += __expf(acc[mi][ni][e] * INV_T);

#pragma unroll
    for (int off = 16; off > 0; off >>= 1)
        s += __shfl_xor(s, off, 32);

    if (lane == 0)
        atomicAdd(accum, (double)s);
}

__global__ void zero_accum(double* acc) {
    acc[0] = 0.0;
    acc[1] = 0.0;
}

__global__ void finalize(const double* __restrict__ acc, float* __restrict__ out) {
    const double pos = acc[0];
    const double neg = acc[1];
    out[0] = (float)(-log(pos / (neg + pos)));
}

// ---------------------------------------------------------------------------
extern "C" void kernel_launch(void* const* d_in, const int* in_sizes, int n_in,
                              void* d_out, int out_size, void* d_ws, size_t ws_size,
                              hipStream_t stream) {
    (void)in_sizes; (void)n_in; (void)out_size; (void)ws_size;

    const float* z_i = (const float*)d_in[0];
    const float* z_j = (const float*)d_in[1];
    float* out = (float*)d_out;

    char* ws = (char*)d_ws;
    const size_t halfBytes = (size_t)N_ROWS * KDIM * sizeof(half_t);  // 8 MiB
    half_t* zi_h = (half_t*)ws;
    half_t* zj_h = (half_t*)(ws + halfBytes);
    double* acc  = (double*)(ws + 2 * halfBytes);

    zero_accum<<<1, 1, 0, stream>>>(acc);

    nrm_rows_f16<<<N_ROWS / 8, 256, 0, stream>>>(z_i, zi_h);
    nrm_rows_f16<<<N_ROWS / 8, 256, 0, stream>>>(z_j, zj_h);

    dim3 grid(N_ROWS / BM, N_ROWS / BN);   // 64 x 32 blocks
    simexp_gemm<<<grid, 256, 0, stream>>>(zi_h, zi_h, &acc[0]);  // pos_sum
    simexp_gemm<<<grid, 256, 0, stream>>>(zi_h, zj_h, &acc[1]);  // neg_sum

    finalize<<<1, 1, 0, stream>>>(acc, out);
}
